// RecurrentRGCN_12824772346299
// MI455X (gfx1250) — matrix operand order repaired and checked by reading, not compiled, not run on hardware
//
#include <hip/hip_runtime.h>

#define N_ENTS  50000
#define NUM_REL 460
#define DD      256
#define TT      8
#define EE      150000
#define GG      300000   /* 2*E */
#define SLOPE_C 0.22916667f
#define LDS_STRIDE 272   /* padded bf16 row stride for A tile in LDS */

typedef __attribute__((ext_vector_type(16))) __bf16 v16bf;
typedef __attribute__((ext_vector_type(8)))  __bf16 v8bf;
typedef __attribute__((ext_vector_type(8)))  float  v8f;

__device__ __forceinline__ __bf16 f2bf(float f) {
  unsigned u = __builtin_bit_cast(unsigned, f);
  unsigned r = u + 0x7FFFu + ((u >> 16) & 1u);   // round-to-nearest-even
  unsigned short hh = (unsigned short)(r >> 16);
  return __builtin_bit_cast(__bf16, hh);
}
__device__ __forceinline__ float sigm(float x) { return 1.f / (1.f + expf(-x)); }

// A fragment (16x32 bf16): lane holds M=lane&15; elems 0..7 -> K=k0+(lane>>4)*8..,
// elems 8..15 -> K=k0+16+(lane>>4)*8.. (per CDNA5 ISA 16-bit A layout)
__device__ __forceinline__ v16bf load_a_g(const __bf16* A, size_t row, int k0, int half) {
  const __bf16* p = A + row * DD + k0 + half * 8;
  v8bf lo = *(const v8bf*)p;
  v8bf hi = *(const v8bf*)(p + 16);
  return __builtin_shufflevector(lo, hi, 0,1,2,3,4,5,6,7,8,9,10,11,12,13,14,15);
}
__device__ __forceinline__ v16bf load_a_lds(const __bf16* As, int m, int k0, int half) {
  const __bf16* p = As + m * LDS_STRIDE + k0 + half * 8;
  v8bf lo = *(const v8bf*)p;
  v8bf hi = *(const v8bf*)(p + 16);
  return __builtin_shufflevector(lo, hi, 0,1,2,3,4,5,6,7,8,9,10,11,12,13,14,15);
}
// B fragment (32x16 bf16) from Bt[n][k] (ld=DD): lane N=n0+(lane&15), elems 0..7 -> K=k0+(lane>>4)*8.., elems 8..15 -> K=k0+16+(lane>>4)*8..
__device__ __forceinline__ v16bf load_b_g(const __bf16* Bt, int n0, int k0, int lane) {
  const __bf16* p = Bt + (size_t)(n0 + (lane & 15)) * DD + k0 + (lane >> 4) * 8;
  v8bf lo = *(const v8bf*)p;
  v8bf hi = *(const v8bf*)(p + 16);
  return __builtin_shufflevector(lo, hi, 0,1,2,3,4,5,6,7,8,9,10,11,12,13,14,15);
}
#define WMMA_BF16(A_, B_, C_) \
  __builtin_amdgcn_wmma_f32_16x16x32_bf16(false, (A_), false, (B_), (short)0, (C_), false, false)

// ---------------- utility kernels ----------------
__global__ __launch_bounds__(256) void k_zero(float* p, long n) {
  long i = (long)blockIdx.x * 256 + threadIdx.x;
  if (i < n) p[i] = 0.f;
}
__global__ __launch_bounds__(256) void k_copy(const float* s, float* d, long n) {
  long i = (long)blockIdx.x * 256 + threadIdx.x;
  if (i < n) d[i] = s[i];
}
// transpose-convert DxD fp32 -> bf16 Bt[n][k] = W[k][n]
__global__ __launch_bounds__(256) void k_convT(const float* W, __bf16* T) {
  int idx = blockIdx.x * 256 + threadIdx.x;   // 65536 total
  int n = idx >> 8, k = idx & 255;
  T[idx] = f2bf(W[k * DD + n]);
}
__global__ __launch_bounds__(256) void k_conv(const float* W, __bf16* T, long n) {
  long i = (long)blockIdx.x * 256 + threadIdx.x;
  if (i < n) T[i] = f2bf(W[i]);
}
// row L2 normalize (block = one row of 256), writes fp32 + bf16 shadow
__global__ __launch_bounds__(256) void k_l2n(const float* src, float* dstF, __bf16* dstB) {
  __shared__ float ssq;
  size_t row = blockIdx.x; int j = threadIdx.x;
  if (j == 0) ssq = 0.f;
  __syncthreads();
  float v = src[row * DD + j];
  atomicAdd(&ssq, v * v);
  __syncthreads();
  float inv = 1.f / fmaxf(sqrtf(ssq), 1e-12f);
  float o = v * inv;
  dstF[row * DD + j] = o;
  dstB[row * DD + j] = f2bf(o);
}

// ---------------- relation evolution ----------------
__global__ __launch_bounds__(256) void k_rel_scatter(const float* __restrict__ h,
                                                     const int* __restrict__ r2e,
                                                     const int* __restrict__ rid,
                                                     float* __restrict__ ssum,
                                                     float* __restrict__ cnt) {
  int i = blockIdx.x;                     // one gathered row per block
  int e = r2e[i]; int r = rid[i];
  int j = threadIdx.x;
  atomicAdd(ssum + (size_t)r * DD + j, h[(size_t)e * DD + j]);
  if (j == 0) atomicAdd(cnt + r, 1.f);
}

__global__ __launch_bounds__(256) void k_rel_gru(const float* __restrict__ emb_rel,
                                                 const float* __restrict__ ssum,
                                                 const float* __restrict__ cnt,
                                                 const float* __restrict__ wih,
                                                 const float* __restrict__ whh,
                                                 const float* __restrict__ bih,
                                                 const float* __restrict__ bhh,
                                                 float* __restrict__ h0,   // in/out
                                                 __bf16* __restrict__ h0b) {
  __shared__ float xin[2 * DD];
  __shared__ float hrow[DD];
  __shared__ float ssq;
  int r = blockIdx.x, j = threadIdx.x;
  float c = cnt[r];
  xin[j]      = emb_rel[(size_t)r * DD + j];
  xin[DD + j] = (c > 0.f) ? ssum[(size_t)r * DD + j] / fmaxf(c, 1.f) : 0.f;
  hrow[j]     = h0[(size_t)r * DD + j];
  if (j == 0) ssq = 0.f;
  __syncthreads();
  const float* wr = wih + (size_t)j * 512;
  const float* wz = wih + (size_t)(256 + j) * 512;
  const float* wg = wih + (size_t)(512 + j) * 512;
  float ir = bih[j], iz = bih[256 + j], ig = bih[512 + j];
  for (int k = 0; k < 512; ++k) { float x = xin[k]; ir += x * wr[k]; iz += x * wz[k]; ig += x * wg[k]; }
  const float* vr = whh + (size_t)j * 256;
  const float* vz = whh + (size_t)(256 + j) * 256;
  const float* vg = whh + (size_t)(512 + j) * 256;
  float hr = bhh[j], hz = bhh[256 + j], hg = bhh[512 + j];
  for (int k = 0; k < 256; ++k) { float x = hrow[k]; hr += x * vr[k]; hz += x * vz[k]; hg += x * vg[k]; }
  float rr = sigm(ir + hr), zz = sigm(iz + hz), gg = tanhf(ig + rr * hg);
  float nh = (1.f - zz) * gg + zz * hrow[j];
  atomicAdd(&ssq, nh * nh);
  __syncthreads();
  float inv = 1.f / fmaxf(sqrtf(ssq), 1e-12f);
  float o = nh * inv;
  h0[(size_t)r * DD + j]  = o;
  h0b[(size_t)r * DD + j] = f2bf(o);
}

__global__ __launch_bounds__(256) void k_deg(const int* __restrict__ dst, float* __restrict__ deg) {
  int i = blockIdx.x * 256 + threadIdx.x;
  if (i < EE) atomicAdd(deg + dst[i], 1.f);
}

// ---------------- RGCN edge message: (cur[src]+h0[et]) @ Wn, scatter-add by dst ----------------
__global__ __launch_bounds__(256) void k_edge_msg(const float* __restrict__ curF,
                                                  const float* __restrict__ h0F,
                                                  const __bf16* __restrict__ wT,
                                                  const int* __restrict__ src,
                                                  const int* __restrict__ et,
                                                  const int* __restrict__ dst,
                                                  float* __restrict__ agg) {
  __shared__ __bf16 As[16 * LDS_STRIDE];
  __shared__ int dsts[16];
  int tile = blockIdx.x, tid = threadIdx.x;
  int r  = tid >> 4;             // edge row within tile
  int kc = (tid & 15) * 16;      // 16-wide K chunk
  int e  = tile * 16 + r;
  if (tid < 16) dsts[tid] = (tile * 16 + tid < EE) ? dst[tile * 16 + tid] : -1;
  if (e < EE) {
    int s = src[e], t_ = et[e];
    const float* a = curF + (size_t)s  * DD + kc;
    const float* b = h0F  + (size_t)t_ * DD + kc;
    #pragma unroll
    for (int i = 0; i < 16; ++i) As[r * LDS_STRIDE + kc + i] = f2bf(a[i] + b[i]);
  } else {
    #pragma unroll
    for (int i = 0; i < 16; ++i) As[r * LDS_STRIDE + kc + i] = f2bf(0.f);
  }
  __syncthreads();
  int w = tid >> 5, lane = tid & 31, m = lane & 15, half = lane >> 4;
  #pragma unroll
  for (int nt = 0; nt < 2; ++nt) {
    int n0 = w * 16 + nt * 128;
    v8f acc = {};
    #pragma unroll
    for (int k0 = 0; k0 < DD; k0 += 32) {
      v16bf a = load_a_lds(As, m, k0, half);
      v16bf b = load_b_g(wT, n0, k0, lane);
      acc = WMMA_BF16(a, b, acc);
    }
    int n = n0 + (lane & 15);
    #pragma unroll
    for (int v = 0; v < 8; ++v) {
      int mm = v + 8 * half;
      int d = dsts[mm];
      if (d >= 0) atomicAdd(agg + (size_t)d * DD + n, acc[v]);
    }
  }
}

// ---------------- self-loop GEMMs + combine (agg*norm + select(loop/evolve), leaky relu) ----------------
__global__ __launch_bounds__(256) void k_loop_combine(const __bf16* __restrict__ curB,
                                                      const __bf16* __restrict__ wlT,
                                                      const __bf16* __restrict__ weT,
                                                      const float* __restrict__ agg,
                                                      const float* __restrict__ deg,
                                                      float* __restrict__ outF,
                                                      __bf16* __restrict__ outB) {
  int tile = blockIdx.x, tid = threadIdx.x;
  int w = tid >> 5, lane = tid & 31, half = lane >> 4;
  size_t rowA = (size_t)tile * 16 + (lane & 15);
  #pragma unroll
  for (int nt = 0; nt < 2; ++nt) {
    int n0 = w * 16 + nt * 128;
    v8f c1 = {}, c2 = {};
    #pragma unroll
    for (int k0 = 0; k0 < DD; k0 += 32) {
      v16bf a  = load_a_g(curB, rowA, k0, half);
      v16bf b1 = load_b_g(wlT, n0, k0, lane);
      v16bf b2 = load_b_g(weT, n0, k0, lane);
      c1 = WMMA_BF16(a, b1, c1);
      c2 = WMMA_BF16(a, b2, c2);
    }
    int n = n0 + (lane & 15);
    #pragma unroll
    for (int v = 0; v < 8; ++v) {
      int mm = v + 8 * half;
      size_t row = (size_t)tile * 16 + mm;
      float d   = deg[row];
      float sel = (d > 0.f) ? c1[v] : c2[v];
      float nrm = (d > 0.f) ? (1.f / d) : 0.f;
      float val = agg[row * DD + n] * nrm + sel;
      val = (val > 0.f) ? val : val * SLOPE_C;
      outF[row * DD + n] = val;
      outB[row * DD + n] = f2bf(val);
    }
  }
}

// ---------------- entity GRU fully fused: gi/gh GEMMs + gates + row L2 norm ----------------
__global__ __launch_bounds__(512) void k_egru(const __bf16* __restrict__ curnB,
                                              const __bf16* __restrict__ hB,
                                              const float* __restrict__ hF,
                                              const __bf16* __restrict__ wihB,
                                              const __bf16* __restrict__ whhB,
                                              const float* __restrict__ bih,
                                              const float* __restrict__ bhh,
                                              float* __restrict__ hOutF,
                                              __bf16* __restrict__ hOutB,
                                              float* __restrict__ hist) {
  __shared__ float rowsum[16];
  int tid = threadIdx.x, tile = blockIdx.x;
  if (tid < 16) rowsum[tid] = 0.f;
  __syncthreads();
  int w = tid >> 5, lane = tid & 31, half = lane >> 4;
  int j0 = w * 16;                              // wave's 16-column gate slice
  size_t rowA = (size_t)tile * 16 + (lane & 15);
  v8f ir = {}, iz = {}, ig = {}, hr = {}, hz = {}, hg = {};
  #pragma unroll
  for (int k0 = 0; k0 < DD; k0 += 32) {
    v16bf a1 = load_a_g(curnB, rowA, k0, half);
    v16bf a2 = load_a_g(hB,    rowA, k0, half);
    v16bf b;
    b = load_b_g(wihB, j0,        k0, lane); ir = WMMA_BF16(a1, b, ir);
    b = load_b_g(wihB, 256 + j0,  k0, lane); iz = WMMA_BF16(a1, b, iz);
    b = load_b_g(wihB, 512 + j0,  k0, lane); ig = WMMA_BF16(a1, b, ig);
    b = load_b_g(whhB, j0,        k0, lane); hr = WMMA_BF16(a2, b, hr);
    b = load_b_g(whhB, 256 + j0,  k0, lane); hz = WMMA_BF16(a2, b, hz);
    b = load_b_g(whhB, 512 + j0,  k0, lane); hg = WMMA_BF16(a2, b, hg);
  }
  int n = j0 + (lane & 15);
  float bir = bih[n], biz = bih[256 + n], big = bih[512 + n];
  float bhr = bhh[n], bhz = bhh[256 + n], bhg = bhh[512 + n];
  float nh[8];
  #pragma unroll
  for (int v = 0; v < 8; ++v) {
    int mm = v + 8 * half;
    size_t row = (size_t)tile * 16 + mm;
    float r = sigm(ir[v] + bir + hr[v] + bhr);
    float z = sigm(iz[v] + biz + hz[v] + bhz);
    float g = tanhf(ig[v] + big + r * (hg[v] + bhg));
    float hv = hF[row * DD + n];
    float o = (1.f - z) * g + z * hv;
    nh[v] = o;
    atomicAdd(&rowsum[mm], o * o);
  }
  __syncthreads();
  #pragma unroll
  for (int v = 0; v < 8; ++v) {
    int mm = v + 8 * half;
    size_t row = (size_t)tile * 16 + mm;
    float inv = 1.f / fmaxf(sqrtf(rowsum[mm]), 1e-12f);
    float o = nh[v] * inv;
    hOutF[row * DD + n] = o;
    hOutB[row * DD + n] = f2bf(o);
    hist[row * DD + n]  = o;
  }
}

// ---------------- host ----------------
extern "C" void kernel_launch(void* const* d_in, const int* in_sizes, int n_in,
                              void* d_out, int out_size, void* d_ws, size_t ws_size,
                              hipStream_t stream) {
  (void)in_sizes; (void)n_in; (void)out_size; (void)ws_size;
  const float* dyn     = (const float*)d_in[0];
  const float* emb_rel = (const float*)d_in[1];
  const float* w_neigh = (const float*)d_in[2];
  const float* w_loop  = (const float*)d_in[3];
  const float* w_evolve= (const float*)d_in[4];
  const float* rwih    = (const float*)d_in[5];
  const float* rwhh    = (const float*)d_in[6];
  const float* rbih    = (const float*)d_in[7];
  const float* rbhh    = (const float*)d_in[8];
  const float* ewih    = (const float*)d_in[9];
  const float* ewhh    = (const float*)d_in[10];
  const float* ebih    = (const float*)d_in[11];
  const float* ebhh    = (const float*)d_in[12];
  const int* src   = (const int*)d_in[13];
  const int* dst   = (const int*)d_in[14];
  const int* etype = (const int*)d_in[15];
  const int* r2e   = (const int*)d_in[16];
  const int* rids  = (const int*)d_in[17];
  float* out = (float*)d_out;

  char* ws = (char*)d_ws;
  size_t off = 0;
  auto carve = [&](size_t bytes) -> char* {
    char* p = ws + off;
    off = (off + bytes + 255) & ~(size_t)255;
    return p;
  };
  float*  h    = (float*) carve((size_t)N_ENTS * DD * 4);
  __bf16* hb   = (__bf16*)carve((size_t)N_ENTS * DD * 2);
  float*  cur  = (float*) carve((size_t)N_ENTS * DD * 4);
  __bf16* curb = (__bf16*)carve((size_t)N_ENTS * DD * 2);
  float*  agg  = (float*) carve((size_t)N_ENTS * DD * 4);
  float*  h0   = (float*) carve((size_t)NUM_REL * DD * 4);
  __bf16* h0b  = (__bf16*)carve((size_t)NUM_REL * DD * 2);
  float*  ssum = (float*) carve((size_t)NUM_REL * DD * 4 + NUM_REL * 4); // cnt appended
  float*  cnt  = ssum + (size_t)NUM_REL * DD;
  float*  deg  = (float*) carve((size_t)N_ENTS * 4);
  __bf16* wnT  = (__bf16*)carve((size_t)2 * DD * DD * 2);
  __bf16* wlT  = (__bf16*)carve((size_t)2 * DD * DD * 2);
  __bf16* weT  = (__bf16*)carve((size_t)2 * DD * DD * 2);
  __bf16* wihB = (__bf16*)carve((size_t)3 * DD * DD * 2);
  __bf16* whhB = (__bf16*)carve((size_t)3 * DD * DD * 2);

  // init: h = l2n(dynamic_emb) (+bf16 shadow), h0 = emb_rel, weights -> bf16
  k_l2n<<<N_ENTS, 256, 0, stream>>>(dyn, h, hb);
  k_copy<<<NUM_REL, 256, 0, stream>>>(emb_rel, h0, (long)NUM_REL * DD);
  for (int l = 0; l < 2; ++l) {
    k_convT<<<256, 256, 0, stream>>>(w_neigh + (size_t)l * DD * DD, wnT + (size_t)l * DD * DD);
    k_convT<<<256, 256, 0, stream>>>(w_loop  + (size_t)l * DD * DD, wlT + (size_t)l * DD * DD);
    k_convT<<<256, 256, 0, stream>>>(w_evolve+ (size_t)l * DD * DD, weT + (size_t)l * DD * DD);
  }
  k_conv<<<768, 256, 0, stream>>>(ewih, wihB, (long)3 * DD * DD);
  k_conv<<<768, 256, 0, stream>>>(ewhh, whhB, (long)3 * DD * DD);

  const long aggN = (long)N_ENTS * DD;
  for (int t = 0; t < TT; ++t) {
    const int* src_t = src   + (size_t)t * EE;
    const int* dst_t = dst   + (size_t)t * EE;
    const int* et_t  = etype + (size_t)t * EE;
    const int* r2e_t = r2e   + (size_t)t * GG;
    const int* rid_t = rids  + (size_t)t * GG;

    // relation evolution
    long zrel = (long)NUM_REL * DD + NUM_REL;
    k_zero<<<(unsigned)((zrel + 255) / 256), 256, 0, stream>>>(ssum, zrel);
    k_rel_scatter<<<GG, 256, 0, stream>>>(h, r2e_t, rid_t, ssum, cnt);
    k_rel_gru<<<NUM_REL, 256, 0, stream>>>(emb_rel, ssum, cnt, rwih, rwhh, rbih, rbhh, h0, h0b);

    // degrees
    k_zero<<<(unsigned)((N_ENTS + 255) / 256), 256, 0, stream>>>(deg, N_ENTS);
    k_deg<<<(EE + 255) / 256, 256, 0, stream>>>(dst_t, deg);

    // RGCN layer 0 (input = h)
    k_zero<<<(unsigned)((aggN + 255) / 256), 256, 0, stream>>>(agg, aggN);
    k_edge_msg<<<(EE + 15) / 16, 256, 0, stream>>>(h, h0, wnT, src_t, et_t, dst_t, agg);
    k_loop_combine<<<N_ENTS / 16, 256, 0, stream>>>(hb, wlT, weT, agg, deg, cur, curb);

    // RGCN layer 1 (input = cur)
    k_zero<<<(unsigned)((aggN + 255) / 256), 256, 0, stream>>>(agg, aggN);
    k_edge_msg<<<(EE + 15) / 16, 256, 0, stream>>>(cur, h0, wnT + (size_t)DD * DD,
                                                   src_t, et_t, dst_t, agg);
    k_loop_combine<<<N_ENTS / 16, 256, 0, stream>>>(curb, wlT + (size_t)DD * DD,
                                                    weT + (size_t)DD * DD, agg, deg, cur, curb);

    // cur = l2n(cur); h = l2n(gru(cur, h)) -> also hist[t]
    k_l2n<<<N_ENTS, 256, 0, stream>>>(cur, cur, curb);
    k_egru<<<N_ENTS / 16, 512, 0, stream>>>(curb, hb, h, wihB, whhB, ebih, ebhh,
                                            h, hb, out + (size_t)t * N_ENTS * DD);
  }
  // h0_fin appended after hist
  k_copy<<<NUM_REL, 256, 0, stream>>>(h0, out + (size_t)TT * N_ENTS * DD, (long)NUM_REL * DD);
}
